// RegularGroupedSlidingAttention_13941463843304
// MI455X (gfx1250) — matrix operand order repaired and checked by reading, not compiled, not run on hardware
//
#include <hip/hip_runtime.h>
#include <hip/hip_bf16.h>

#define B_   2
#define S_   2048
#define D_   2048
#define H_   16
#define KVH_ 4
#define DH_  128
#define WIN_ 128

typedef __bf16 bf16;
typedef __attribute__((ext_vector_type(16))) __bf16 bf16x16;
typedef __attribute__((ext_vector_type(4)))  __bf16 bf16x4;
typedef __attribute__((ext_vector_type(8)))  float  f32x8;
typedef __attribute__((ext_vector_type(4)))  int    i32x4;

// ---------------------------------------------------------------------------
// CDNA5 async global->LDS copy (ASYNCcnt path), probed via __has_builtin with
// a synchronous fallback so the build can't break if the builtin is absent.
// Probe round 2 showed the builtin exists with int4-typed AS pointers.
// ---------------------------------------------------------------------------
#if defined(__has_builtin)
#  if __has_builtin(__builtin_amdgcn_global_load_async_to_lds_b128)
#    define USE_ASYNC_LDS 1
#  endif
#endif
#ifndef USE_ASYNC_LDS
#  define USE_ASYNC_LDS 0
#endif

#define AS1 __attribute__((address_space(1)))
#define AS3 __attribute__((address_space(3)))

__device__ __forceinline__ void cp16_to_lds(const bf16* g, bf16* l) {
#if USE_ASYNC_LDS
  // ISA 10.2: flat LDS address low 32 bits are the LDS byte offset.
  __builtin_amdgcn_global_load_async_to_lds_b128(
      (AS1 i32x4*)(uintptr_t)g,
      (AS3 i32x4*)(uint32_t)(uintptr_t)l,
      0, 0);
#else
  *reinterpret_cast<uint4*>(l) = *reinterpret_cast<const uint4*>(g);
#endif
}

__device__ __forceinline__ void wait_async0() {
#if USE_ASYNC_LDS
#if defined(__has_builtin) && __has_builtin(__builtin_amdgcn_s_wait_asynccnt)
  __builtin_amdgcn_s_wait_asynccnt(0);
#else
  asm volatile("s_wait_asynccnt 0x0" ::: "memory");
#endif
#endif
}

// ---------------------------------------------------------------------------
// WMMA fragment loaders (wave32; A and B both use k = 16*half + e)
// ---------------------------------------------------------------------------
__device__ __forceinline__ bf16x16 load_frag_a(const bf16* lds, int ld, int row0, int k0) {
  const int lane = threadIdx.x & 31;
  const int m = row0 + (lane & 15);
  const int half = lane >> 4;
  bf16x16 f;
#pragma unroll
  for (int e = 0; e < 16; ++e) {
    int kk = k0 + e + half * 16;
    f[e] = lds[m * ld + kk];
  }
  return f;
}

// B fragment: logical [K=32 x N=16]; element e -> K = e + half*16, N = lane&15.
__device__ __forceinline__ bf16x16 load_frag_b_nk(const bf16* lds, int ld, int n0, int k0) {
  const int lane = threadIdx.x & 31;
  const int n = n0 + (lane & 15);
  const int half = lane >> 4;
  bf16x16 f;
#pragma unroll
  for (int e = 0; e < 16; ++e) {
    int kk = k0 + e + half * 16;
    f[e] = lds[n * ld + kk];
  }
  return f;
}

__device__ __forceinline__ bf16x16 load_frag_b_kn(const bf16* lds, int ld, int k0, int n0) {
  const int lane = threadIdx.x & 31;
  const int n = n0 + (lane & 15);
  const int half = lane >> 4;
  bf16x16 f;
#pragma unroll
  for (int e = 0; e < 16; ++e) {
    int kk = k0 + e + half * 16;
    f[e] = lds[kk * ld + n];
  }
  return f;
}

// ---------------------------------------------------------------------------
// fp32 -> bf16 conversion
// ---------------------------------------------------------------------------
__global__ void cvt_f32_bf16_kernel(const float* __restrict__ src,
                                    bf16* __restrict__ dst, size_t n) {
  size_t i = ((size_t)blockIdx.x * blockDim.x + threadIdx.x) * 4;
  if (i + 3 < n) {
    float4 v = *reinterpret_cast<const float4*>(src + i);
    bf16x4 o;
    o[0] = (bf16)v.x; o[1] = (bf16)v.y; o[2] = (bf16)v.z; o[3] = (bf16)v.w;
    *reinterpret_cast<bf16x4*>(dst + i) = o;
  } else {
    for (; i < n; ++i) dst[i] = (bf16)src[i];
  }
}

// ---------------------------------------------------------------------------
// bf16 GEMM: C[M,N] = A[M,K] @ W[N,K]^T
// 128x128 block tile, 8 waves, double-buffered LDS, one barrier per K-step,
// async global->LDS copies overlap WMMA on the other buffer.
// ---------------------------------------------------------------------------
template <typename OutT>
__global__ __launch_bounds__(256) void gemm_bf16_tn(
    const bf16* __restrict__ A, const bf16* __restrict__ W,
    OutT* __restrict__ C, int M, int N, int K) {
  constexpr int LDT = 40;  // 80 B/row: 16B-aligned rows, conflict-free stride
  __shared__ bf16 At[2][128 * LDT];
  __shared__ bf16 Bt[2][128 * LDT];

  const int tid = threadIdx.x;
  const int wave = tid >> 5;
  const int lane = tid & 31;
  const int half = lane >> 4;
  const int nidx = lane & 15;
  const int n0blk = blockIdx.x * 128;
  const int m0blk = blockIdx.y * 128;
  const int wm = (wave >> 1) * 32;
  const int wn = (wave & 1) * 64;

  auto issue_tile = [&](int k0, int buf) {
#pragma unroll
    for (int c = 0; c < 2; ++c) {
      int idx = tid + c * 256;       // 0..511
      int row = idx >> 2;            // 0..127
      int seg = (idx & 3) * 8;       // 0,8,16,24
      cp16_to_lds(&A[(size_t)(m0blk + row) * K + k0 + seg], &At[buf][row * LDT + seg]);
      cp16_to_lds(&W[(size_t)(n0blk + row) * K + k0 + seg], &Bt[buf][row * LDT + seg]);
    }
  };

  f32x8 acc[2][4];
#pragma unroll
  for (int i = 0; i < 2; ++i)
#pragma unroll
    for (int j = 0; j < 4; ++j)
#pragma unroll
      for (int e = 0; e < 8; ++e) acc[i][j][e] = 0.f;

  issue_tile(0, 0);
  const int T = K / 32;
  for (int t = 0; t < T; ++t) {
    wait_async0();
    __syncthreads();  // tile t resident; everyone done with buffer t&1's prior use
    if (t + 1 < T) issue_tile((t + 1) * 32, (t + 1) & 1);

    const bf16* Ab = At[t & 1];
    const bf16* Bb = Bt[t & 1];
    bf16x16 af[2], bw[4];
#pragma unroll
    for (int i = 0; i < 2; ++i) af[i] = load_frag_a(Ab, LDT, wm + i * 16, 0);
#pragma unroll
    for (int j = 0; j < 4; ++j) bw[j] = load_frag_b_nk(Bb, LDT, wn + j * 16, 0);
#pragma unroll
    for (int i = 0; i < 2; ++i)
#pragma unroll
      for (int j = 0; j < 4; ++j)
        acc[i][j] = __builtin_amdgcn_wmma_f32_16x16x32_bf16(
            false, af[i], false, bw[j], (short)0, acc[i][j], false, false);
  }

#pragma unroll
  for (int i = 0; i < 2; ++i)
#pragma unroll
    for (int j = 0; j < 4; ++j)
#pragma unroll
      for (int r = 0; r < 8; ++r) {
        int gm = m0blk + wm + i * 16 + r + half * 8;
        int gn = n0blk + wn + j * 16 + nidx;
        C[(size_t)gm * N + gn] = (OutT)acc[i][j][r];
      }
}

// ---------------------------------------------------------------------------
// RoPE (repeat_interleave cos/sin + rotate_half, theta=10000), in-place bf16
// ---------------------------------------------------------------------------
__global__ void rope_kernel(bf16* __restrict__ t, int NH, int total) {
  int i = blockIdx.x * blockDim.x + threadIdx.x;
  if (i >= total) return;
  int d = i & 63;
  int s = (i / (64 * NH)) % S_;
  size_t base = ((size_t)(i >> 6)) * DH_;  // flattened (b,s,h)

  float x1 = (float)t[base + d];
  float x2 = (float)t[base + d + 64];
  const float L64 = 13.287712379549449f / 64.f;  // log2(10000)/64
  float j1 = (float)(d >> 1);
  float a1 = (float)s * exp2f(-j1 * L64);
  float a2 = (float)s * exp2f(-(32.f + j1) * L64);
  float s1, c1, s2, c2;
  __sincosf(a1, &s1, &c1);
  __sincosf(a2, &s2, &c2);
  t[base + d]      = (bf16)(x1 * c1 - x2 * s1);
  t[base + d + 64] = (bf16)(x2 * c2 + x1 * s2);
}

// ---------------------------------------------------------------------------
// Flash attention, mask: valid iff kv <= q + WIN. 64-q tile / block, 4 waves.
// Double-buffered K/V tiles with async global->LDS streaming.
// ---------------------------------------------------------------------------
__global__ __launch_bounds__(128) void flash_attn_kernel(
    const bf16* __restrict__ Q,   // [B,S,H,DH]
    const bf16* __restrict__ Kb,  // [B,S,KVH,DH]
    const bf16* __restrict__ Vb,  // [B,S,KVH,DH]
    bf16* __restrict__ O) {       // [B,S,H,DH]
  constexpr int LDQ = 136;  // 272 B/row, 16B aligned
  constexpr int LDP = 40;
  __shared__ bf16 Qt[64 * LDQ];
  __shared__ bf16 Kt[2][32 * LDQ];
  __shared__ bf16 Vt[2][32 * LDQ];
  __shared__ bf16 Pt[4 * 16 * LDP];

  const int tid = threadIdx.x;
  const int wave = tid >> 5;
  const int lane = tid & 31;
  const int half = lane >> 4;
  const int nidx = lane & 15;

  const int qt = blockIdx.x % (S_ / 64);
  const int h  = (blockIdx.x / (S_ / 64)) % H_;
  const int b  = blockIdx.x / ((S_ / 64) * H_);
  const int kvh = h / (H_ / KVH_);
  const int q0 = qt * 64;

  const bf16* ksrc = Kb + (((size_t)b * S_) * KVH_ + kvh) * DH_;
  const bf16* vsrc = Vb + (((size_t)b * S_) * KVH_ + kvh) * DH_;

  auto issue_kv = [&](int kv0, int buf) {
#pragma unroll
    for (int c = 0; c < 4; ++c) {
      int idx = tid + c * 128;
      int row = idx >> 4;  // 0..31
      int seg = (idx & 15) * 8;
      size_t goff = (size_t)(kv0 + row) * KVH_ * DH_ + seg;
      cp16_to_lds(&ksrc[goff], &Kt[buf][row * LDQ + seg]);
      cp16_to_lds(&vsrc[goff], &Vt[buf][row * LDQ + seg]);
    }
  };

  {  // load Q tile 64x128 (one-time)
    const bf16* qsrc = Q + (((size_t)b * S_ + q0) * H_ + h) * DH_;
#pragma unroll
    for (int c = 0; c < 8; ++c) {
      int idx = tid + c * 128;
      int row = idx >> 4;
      int seg = (idx & 15) * 8;
      uint4 v = *reinterpret_cast<const uint4*>(&qsrc[(size_t)row * H_ * DH_ + seg]);
      *reinterpret_cast<uint4*>(&Qt[row * LDQ + seg]) = v;
    }
  }
  issue_kv(0, 0);  // prefetch first K/V tile while Q frags are prepared
  __syncthreads();

  bf16x16 qf[4];
#pragma unroll
  for (int kk = 0; kk < 4; ++kk) qf[kk] = load_frag_a(Qt, LDQ, wave * 16, kk * 32);

  f32x8 acc[8];
#pragma unroll
  for (int jd = 0; jd < 8; ++jd)
#pragma unroll
    for (int e = 0; e < 8; ++e) acc[jd][e] = 0.f;

  float m_run[8], l_run[8];
#pragma unroll
  for (int r = 0; r < 8; ++r) { m_run[r] = -1e30f; l_run[r] = 0.f; }

  const float SCL = 0.08838834764831845f * 1.4426950408889634f;  // DH^-0.5 * log2e
  const int kv_end = min(S_, q0 + 64 + WIN_);
  const int T = kv_end / 32;

  for (int t = 0; t < T; ++t) {
    const int kv0 = t * 32;
    wait_async0();
    __syncthreads();  // K/V tile t resident; prior reads of buffer t&1 done
    if (t + 1 < T) issue_kv(kv0 + 32, (t + 1) & 1);
    const bf16* Kc = Kt[t & 1];
    const bf16* Vc = Vt[t & 1];

    // S = Q K^T (two 16-col chunks x four 32-deep k-steps)
    f32x8 sf[2];
#pragma unroll
    for (int j = 0; j < 2; ++j) {
#pragma unroll
      for (int e = 0; e < 8; ++e) sf[j][e] = 0.f;
#pragma unroll
      for (int kk = 0; kk < 4; ++kk) {
        bf16x16 kf = load_frag_b_nk(Kc, LDQ, j * 16, kk * 32);
        sf[j] = __builtin_amdgcn_wmma_f32_16x16x32_bf16(
            false, qf[kk], false, kf, (short)0, sf[j], false, false);
      }
    }

    // scale + mask (kv > q + WIN -> -inf)
#pragma unroll
    for (int j = 0; j < 2; ++j) {
      int kvg = kv0 + j * 16 + nidx;
#pragma unroll
      for (int r = 0; r < 8; ++r) {
        int qg = q0 + wave * 16 + r + half * 8;
        float tt = sf[j][r] * SCL;
        if (kvg > qg + WIN_) tt = -__builtin_inff();
        sf[j][r] = tt;
      }
    }

    // online softmax; xor 1..8 stays within each 16-lane half
    float mnew[8];
#pragma unroll
    for (int r = 0; r < 8; ++r) {
      float tt = fmaxf(sf[0][r], sf[1][r]);
#pragma unroll
      for (int o = 8; o >= 1; o >>= 1) tt = fmaxf(tt, __shfl_xor(tt, o, 32));
      mnew[r] = fmaxf(m_run[r], tt);
    }
    float cf[8];
#pragma unroll
    for (int r = 0; r < 8; ++r) {
      cf[r] = exp2f(m_run[r] - mnew[r]);
      m_run[r] = mnew[r];
    }
    float rs[8];
#pragma unroll
    for (int r = 0; r < 8; ++r) rs[r] = 0.f;
#pragma unroll
    for (int j = 0; j < 2; ++j)
#pragma unroll
      for (int r = 0; r < 8; ++r) {
        float p = exp2f(sf[j][r] - m_run[r]);
        sf[j][r] = p;
        rs[r] += p;
      }
#pragma unroll
    for (int r = 0; r < 8; ++r) {
#pragma unroll
      for (int o = 8; o >= 1; o >>= 1) rs[r] += __shfl_xor(rs[r], o, 32);
      l_run[r] = l_run[r] * cf[r] + rs[r];
    }
#pragma unroll
    for (int jd = 0; jd < 8; ++jd)
#pragma unroll
      for (int r = 0; r < 8; ++r) acc[jd][r] *= cf[r];

    // stage P (C-layout -> A-layout) via per-wave LDS; same-wave DS in-order
    bf16* Pw = &Pt[wave * 16 * LDP];
#pragma unroll
    for (int j = 0; j < 2; ++j)
#pragma unroll
      for (int r = 0; r < 8; ++r)
        Pw[(r + half * 8) * LDP + j * 16 + nidx] = (bf16)sf[j][r];
    bf16x16 pf = load_frag_a(Pw, LDP, 0, 0);

    // O += P V (contract over 32 kv, 8 dh chunks)
#pragma unroll
    for (int jd = 0; jd < 8; ++jd) {
      bf16x16 vf = load_frag_b_kn(Vc, LDQ, 0, jd * 16);
      acc[jd] = __builtin_amdgcn_wmma_f32_16x16x32_bf16(
          false, pf, false, vf, (short)0, acc[jd], false, false);
    }
  }

  // normalize + store
  bf16* odst = O + (((size_t)b * S_ + q0 + wave * 16) * H_ + h) * DH_;
#pragma unroll
  for (int jd = 0; jd < 8; ++jd)
#pragma unroll
    for (int r = 0; r < 8; ++r) {
      int m = r + half * 8;
      odst[(size_t)m * H_ * DH_ + jd * 16 + nidx] = (bf16)(acc[jd][r] / l_run[r]);
    }
}

// ---------------------------------------------------------------------------
// launcher
// ---------------------------------------------------------------------------
extern "C" void kernel_launch(void* const* d_in, const int* in_sizes, int n_in,
                              void* d_out, int out_size, void* d_ws, size_t ws_size,
                              hipStream_t stream) {
  const float* x  = (const float*)d_in[0];
  const float* Wq = (const float*)d_in[1];
  const float* Wk = (const float*)d_in[2];
  const float* Wv = (const float*)d_in[3];
  const float* Wo = (const float*)d_in[4];
  float* out = (float*)d_out;

  char* wsp = (char*)d_ws;
  auto alloc = [&](size_t bytes) -> char* {
    char* p = wsp;
    wsp += (bytes + 255) & ~(size_t)255;
    return p;
  };
  bf16* xb  = (bf16*)alloc((size_t)B_ * S_ * D_ * 2);
  bf16* wqb = (bf16*)alloc((size_t)H_ * DH_ * D_ * 2);
  bf16* wkb = (bf16*)alloc((size_t)KVH_ * DH_ * D_ * 2);
  bf16* wvb = (bf16*)alloc((size_t)KVH_ * DH_ * D_ * 2);
  bf16* wob = (bf16*)alloc((size_t)D_ * H_ * DH_ * 2);
  bf16* qb  = (bf16*)alloc((size_t)B_ * S_ * H_ * DH_ * 2);
  bf16* kb  = (bf16*)alloc((size_t)B_ * S_ * KVH_ * DH_ * 2);
  bf16* vb  = (bf16*)alloc((size_t)B_ * S_ * KVH_ * DH_ * 2);
  bf16* ob  = (bf16*)alloc((size_t)B_ * S_ * H_ * DH_ * 2);

  auto cvt = [&](const float* src, bf16* dst, size_t n) {
    int blocks = (int)((n / 4 + 255) / 256);
    cvt_f32_bf16_kernel<<<blocks, 256, 0, stream>>>(src, dst, n);
  };
  cvt(x,  xb,  (size_t)B_ * S_ * D_);
  cvt(Wq, wqb, (size_t)H_ * DH_ * D_);
  cvt(Wk, wkb, (size_t)KVH_ * DH_ * D_);
  cvt(Wv, wvb, (size_t)KVH_ * DH_ * D_);
  cvt(Wo, wob, (size_t)D_ * H_ * DH_);

  const int M = B_ * S_;
  gemm_bf16_tn<bf16><<<dim3((H_ * DH_) / 128, M / 128), 256, 0, stream>>>(
      xb, wqb, qb, M, H_ * DH_, D_);
  gemm_bf16_tn<bf16><<<dim3((KVH_ * DH_) / 128, M / 128), 256, 0, stream>>>(
      xb, wkb, kb, M, KVH_ * DH_, D_);
  gemm_bf16_tn<bf16><<<dim3((KVH_ * DH_) / 128, M / 128), 256, 0, stream>>>(
      xb, wvb, vb, M, KVH_ * DH_, D_);

  int total_q = B_ * S_ * H_ * 64;
  rope_kernel<<<(total_q + 255) / 256, 256, 0, stream>>>(qb, H_, total_q);
  int total_k = B_ * S_ * KVH_ * 64;
  rope_kernel<<<(total_k + 255) / 256, 256, 0, stream>>>(kb, KVH_, total_k);

  flash_attn_kernel<<<B_ * H_ * (S_ / 64), 128, 0, stream>>>(qb, kb, vb, ob);

  gemm_bf16_tn<float><<<dim3(D_ / 128, M / 128), 256, 0, stream>>>(
      ob, wob, out, M, D_, H_ * DH_);
}
